// CSSMTinyViTBlock_24412594110892
// MI455X (gfx1250) — matrix operand order repaired and checked by reading, not compiled, not run on hardware
//
#include <hip/hip_runtime.h>
#include <hip/hip_bf16.h>

// ---------------- types ----------------
typedef __attribute__((ext_vector_type(16))) __bf16 v16bf;
typedef __attribute__((ext_vector_type(8)))  float  v8f;
typedef __attribute__((ext_vector_type(4)))  unsigned int u32x4;

#define CDIM   384
#define HIDDIM 1536
#define TILE_M 32
#define NTOK   16384

// ws layout (ushort element offsets), all 32B aligned
#define OFF_WU 0
#define OFF_WG 147456
#define OFF_A  294912
#define OFF_W1 442368
#define OFF_W2 1032192
#define W_TOTAL 1622016

union Frag16 {
  v16bf v;
  u32x4 q[2];
};

__device__ __forceinline__ unsigned short f2bf(float f) {
  unsigned int u = __float_as_uint(f);
  u += 0x7FFFu + ((u >> 16) & 1u);          // round-to-nearest-even
  return (unsigned short)(u >> 16);
}

__device__ __forceinline__ v8f bcast8(float v) {
  v8f r;
#pragma unroll
  for (int i = 0; i < 8; ++i) r[i] = v;
  return r;
}

// ---- WMMA fragment loaders (CDNA5 16x16x32 bf16 layouts) ----
// A (16x32): lane<16 holds row=lane, K {kb..kb+7, kb+16..kb+23};
//            lane>=16 holds row=lane-16, K {kb+8..kb+15, kb+24..kb+31}
__device__ __forceinline__ v16bf load_a_frag(const unsigned short* buf, int ld,
                                             int rbase, int kbase, int lane) {
  int row = rbase + (lane & 15);
  int k0  = kbase + ((lane >> 4) << 3);
  const unsigned short* p = buf + row * ld + k0;
  Frag16 f;
  f.q[0] = *(const u32x4*)(p);
  f.q[1] = *(const u32x4*)(p + 16);
  return f.v;
}

// B (32x16): lane holds column N=nbase+(lane&15), K {kb+8h..+7, kb+16+8h..+7}, h=lane>>4
__device__ __forceinline__ v16bf load_b_frag(const unsigned short* W, int ld,
                                             int kbase, int nbase, int lane) {
  const unsigned short* p = W + (unsigned)(kbase + ((lane >> 4) << 3)) * ld + nbase + (lane & 15);
  Frag16 f;
#pragma unroll
  for (int j = 0; j < 4; ++j) {
    f.q[0][j] = (unsigned)p[(2 * j) * ld] | ((unsigned)p[(2 * j + 1) * ld] << 16);
    f.q[1][j] = (unsigned)p[(16 + 2 * j) * ld] | ((unsigned)p[(17 + 2 * j) * ld] << 16);
  }
  return f.v;
}

__device__ __forceinline__ v8f wmma_bf16(v16bf a, v16bf b, v8f c) {
  return __builtin_amdgcn_wmma_f32_16x16x32_bf16(false, a, false, b, (short)0, c,
                                                 false, false);
}

// ---------------- weight conversion kernel ----------------
__global__ __launch_bounds__(256) void weights_to_bf16(
    const float* __restrict__ Wu, const float* __restrict__ Wg,
    const float* __restrict__ A,  const float* __restrict__ W1,
    const float* __restrict__ W2, unsigned short* __restrict__ ws) {
  int i = blockIdx.x * 256 + threadIdx.x;
  const int S = 147456, SL = 589824;
  if (i < S)                 ws[i] = f2bf(Wu[i]);
  else if (i < 2 * S)        ws[i] = f2bf(Wg[i - S]);
  else if (i < 3 * S)        ws[i] = f2bf(A[i - 2 * S]);
  else if (i < 3 * S + SL)   ws[i] = f2bf(W1[i - 3 * S]);
  else if (i < W_TOTAL)      ws[i] = f2bf(W2[i - 3 * S - SL]);
}

// ---------------- fully fused CSSM TinyViT block ----------------
// Grid: 512 blocks x 256 threads (8 wave32). Each block = 32 tokens.
// Wave tiling: rt = wave>>2 (16-row tile), cq = wave&3 (96-col quarter).
// launch_bounds(256,4): >=4 waves/SIMD => <=256 VGPRs => 2 workgroups/WGP.
__global__ __launch_bounds__(256, 4) void cssm_block_kernel(
    const float* __restrict__ x,
    const float* __restrict__ n1s, const float* __restrict__ n1b,
    const float* __restrict__ bu,  const float* __restrict__ bg,
    const float* __restrict__ n2s, const float* __restrict__ n2b,
    const float* __restrict__ b1,  const float* __restrict__ b2,
    const unsigned short* __restrict__ wbf,
    float* __restrict__ out) {
  __shared__ __align__(16) unsigned short hA[TILE_M * CDIM];  // 24 KB
  __shared__ __align__(16) unsigned short hB[TILE_M * CDIM];  // 24 KB
  __shared__ float redS[TILE_M * 4];
  __shared__ float redQ[TILE_M * 4];

  const unsigned short* Wu = wbf + OFF_WU;
  const unsigned short* Wg = wbf + OFF_WG;
  const unsigned short* Am = wbf + OFF_A;
  const unsigned short* W1 = wbf + OFF_W1;
  const unsigned short* W2 = wbf + OFF_W2;

  const int lane  = threadIdx.x & 31;
  const int wave  = threadIdx.x >> 5;
  const int rt    = wave >> 2;           // 0..1
  const int cq    = wave & 3;            // 0..3
  const int rbase = rt << 4;             // local row base of wave tile
  const int fcol  = lane & 15;           // N within 16x16 tile
  const int frow  = (lane >> 4) << 3;    // row offset 0 or 8
  const int tok0  = blockIdx.x * TILE_M;

  // ---------------- LayerNorm 1 (x -> bf16 xn in hA) ----------------
  for (int r = 0; r < 4; ++r) {
    int lrow = wave * 4 + r;
    const float* xp = x + (size_t)(tok0 + lrow) * CDIM;
    float v[12], s = 0.f, q = 0.f;
#pragma unroll
    for (int j = 0; j < 12; ++j) {
      v[j] = xp[lane + j * 32];
      s += v[j];
      q = fmaf(v[j], v[j], q);
    }
#pragma unroll
    for (int m = 16; m >= 1; m >>= 1) {
      s += __shfl_xor(s, m, 32);
      q += __shfl_xor(q, m, 32);
    }
    float mu  = s * (1.f / CDIM);
    float var = q * (1.f / CDIM) - mu * mu;
    float rs  = rsqrtf(var + 1e-6f);
#pragma unroll
    for (int j = 0; j < 12; ++j) {
      int c = lane + j * 32;
      hA[lrow * CDIM + c] = f2bf((v[j] - mu) * rs * n1s[c] + n1b[c]);
    }
  }
  __syncthreads();

  // -------- u / g projections (bias seeded into WMMA C, kept in regs) -------
  float ureg[6][8], greg[6][8];
  for (int ct = 0; ct < 6; ++ct) {
    int nb = cq * 96 + ct * 16;
    v8f au = bcast8(bu[nb + fcol]);
    v8f ag = bcast8(bg[nb + fcol]);
    for (int k = 0; k < 12; ++k) {
      v16bf a = load_a_frag(hA, CDIM, rbase, k * 32, lane);
      au = wmma_bf16(a, load_b_frag(Wu, CDIM, k * 32, nb, lane), au);
      ag = wmma_bf16(a, load_b_frag(Wg, CDIM, k * 32, nb, lane), ag);
    }
#pragma unroll
    for (int i = 0; i < 8; ++i) {
      ureg[ct][i] = au[i];
      greg[ct][i] = 1.f / (1.f + __expf(-ag[i]));
    }
  }
  __syncthreads();  // everyone done reading xn from hA

  // ---------------- h1 = (1-g)*u  (h0 = 0) -> hA ----------------
  for (int ct = 0; ct < 6; ++ct) {
    int nb = cq * 96 + ct * 16;
#pragma unroll
    for (int i = 0; i < 8; ++i) {
      float h = (1.f - greg[ct][i]) * ureg[ct][i];
      hA[(rbase + frow + i) * CDIM + nb + fcol] = f2bf(h);
    }
  }
  __syncthreads();

  // ---------------- scan: 7 remaining steps, h double-buffered ----------------
  float xr[6][8];
  for (int s = 0; s < 7; ++s) {
    const unsigned short* src = (s & 1) ? hB : hA;
    unsigned short*       dst = (s & 1) ? hA : hB;
    for (int ct = 0; ct < 6; ++ct) {
      int nb = cq * 96 + ct * 16;
      v8f acc = {};
      for (int k = 0; k < 12; ++k)
        acc = wmma_bf16(load_a_frag(src, CDIM, rbase, k * 32, lane),
                        load_b_frag(Am, CDIM, k * 32, nb, lane), acc);
      if (s < 6) {
#pragma unroll
        for (int i = 0; i < 8; ++i) {
          // g*acc + (1-g)*u == u + g*(acc-u)
          float h = fmaf(greg[ct][i], acc[i] - ureg[ct][i], ureg[ct][i]);
          dst[(rbase + frow + i) * CDIM + nb + fcol] = f2bf(h);
        }
      } else {  // final step: residual xr = x + h_last, stays in registers
#pragma unroll
        for (int i = 0; i < 8; ++i) {
          float h = fmaf(greg[ct][i], acc[i] - ureg[ct][i], ureg[ct][i]);
          int lrow = rbase + frow + i;
          xr[ct][i] = x[(size_t)(tok0 + lrow) * CDIM + nb + fcol] + h;
        }
      }
    }
    if (s < 6) __syncthreads();
  }

  // ---------------- LayerNorm 2 (cross-wave partials via LDS) ----------------
  {
    float ssum[8], ssq[8];
#pragma unroll
    for (int i = 0; i < 8; ++i) {
      float s1 = 0.f, s2 = 0.f;
#pragma unroll
      for (int ct = 0; ct < 6; ++ct) {
        float v = xr[ct][i];
        s1 += v;
        s2 = fmaf(v, v, s2);
      }
#pragma unroll
      for (int m = 8; m >= 1; m >>= 1) {  // reduce within 16-lane half
        s1 += __shfl_xor(s1, m, 32);
        s2 += __shfl_xor(s2, m, 32);
      }
      ssum[i] = s1; ssq[i] = s2;
    }
    if (fcol == 0) {
#pragma unroll
      for (int i = 0; i < 8; ++i) {
        int lrow = rbase + frow + i;
        redS[lrow * 4 + cq] = ssum[i];
        redQ[lrow * 4 + cq] = ssq[i];
      }
    }
  }
  __syncthreads();  // also guarantees all scan reads of hA are done

  // oacc is the MLP accumulator, seeded with the residual and final bias:
  // out = xr + b2 + sum_hc m_chunk @ W2_chunk
  v8f oacc[6];
  {
    float mu8[8], rs8[8];
#pragma unroll
    for (int i = 0; i < 8; ++i) {
      int lrow = rbase + frow + i;
      float s1 = redS[lrow * 4 + 0] + redS[lrow * 4 + 1] +
                 redS[lrow * 4 + 2] + redS[lrow * 4 + 3];
      float s2 = redQ[lrow * 4 + 0] + redQ[lrow * 4 + 1] +
                 redQ[lrow * 4 + 2] + redQ[lrow * 4 + 3];
      float mu = s1 * (1.f / CDIM);
      mu8[i] = mu;
      rs8[i] = rsqrtf(s2 * (1.f / CDIM) - mu * mu + 1e-6f);
    }
    for (int ct = 0; ct < 6; ++ct) {
      int nb = cq * 96 + ct * 16;
      int c  = nb + fcol;
      float sc = n2s[c], bi = n2b[c], bb2 = b2[c];
#pragma unroll
      for (int i = 0; i < 8; ++i) {
        float xn = (xr[ct][i] - mu8[i]) * rs8[i] * sc + bi;
        hA[(rbase + frow + i) * CDIM + c] = f2bf(xn);  // xn2 -> hA
        oacc[ct][i] = xr[ct][i] + bb2;                 // xr dies here
      }
    }
  }
  __syncthreads();

  // ---------------- MLP fused over HID chunks of 128 ----------------
  unsigned short* mbuf = hB;  // 32 x 128 bf16 gelu intermediate
  for (int hc = 0; hc < 12; ++hc) {
    // m_chunk = gelu(xn2 @ W1[:, hc*128 : +128] + b1)
    for (int j = 0; j < 2; ++j) {
      int nbl = cq * 32 + j * 16;  // 0..112 within chunk
      v8f mac = bcast8(b1[hc * 128 + nbl + fcol]);
      for (int k = 0; k < 12; ++k)
        mac = wmma_bf16(load_a_frag(hA, CDIM, rbase, k * 32, lane),
                        load_b_frag(W1, HIDDIM, k * 32, hc * 128 + nbl, lane),
                        mac);
#pragma unroll
      for (int i = 0; i < 8; ++i) {
        float v = mac[i];
        float t = 0.7978845608028654f * fmaf(0.044715f * v, v * v, v);
        float g = 0.5f * v * (1.f + tanhf(t));  // jax.nn.gelu (tanh approx)
        mbuf[(rbase + frow + i) * 128 + nbl + fcol] = f2bf(g);
      }
    }
    __syncthreads();
    // out_acc += m_chunk @ W2[hc*128 : +128, :]
    for (int ct = 0; ct < 6; ++ct) {
      int nb = cq * 96 + ct * 16;
      v8f acc = oacc[ct];
      for (int k = 0; k < 4; ++k)
        acc = wmma_bf16(load_a_frag(mbuf, 128, rbase, k * 32, lane),
                        load_b_frag(W2, CDIM, hc * 128 + k * 32, nb, lane),
                        acc);
      oacc[ct] = acc;
    }
    __syncthreads();  // protect mbuf before next chunk's writes
  }

  // ---------------- store out = oacc (residual + bias already seeded) -------
  for (int ct = 0; ct < 6; ++ct) {
    int nb = cq * 96 + ct * 16;
#pragma unroll
    for (int i = 0; i < 8; ++i) {
      int lrow = rbase + frow + i;
      out[(size_t)(tok0 + lrow) * CDIM + nb + fcol] = oacc[ct][i];
    }
  }
}

// ---------------- host launcher ----------------
extern "C" void kernel_launch(void* const* d_in, const int* in_sizes, int n_in,
                              void* d_out, int out_size, void* d_ws,
                              size_t ws_size, hipStream_t stream) {
  (void)in_sizes; (void)n_in; (void)out_size; (void)ws_size;
  const float* x   = (const float*)d_in[0];
  const float* n1s = (const float*)d_in[1];
  const float* n1b = (const float*)d_in[2];
  const float* Wu  = (const float*)d_in[3];
  const float* bu  = (const float*)d_in[4];
  const float* Wg  = (const float*)d_in[5];
  const float* bg  = (const float*)d_in[6];
  const float* A   = (const float*)d_in[7];
  const float* n2s = (const float*)d_in[8];
  const float* n2b = (const float*)d_in[9];
  const float* W1  = (const float*)d_in[10];
  const float* b1  = (const float*)d_in[11];
  const float* W2  = (const float*)d_in[12];
  const float* b2  = (const float*)d_in[13];
  unsigned short* wbf = (unsigned short*)d_ws;

  weights_to_bf16<<<(W_TOTAL + 255) / 256, 256, 0, stream>>>(Wu, Wg, A, W1, W2,
                                                             wbf);
  cssm_block_kernel<<<NTOK / TILE_M, 256, 0, stream>>>(
      x, n1s, n1b, bu, bg, n2s, n2b, b1, b2, wbf, (float*)d_out);
}
